// Qwen3MoeAttention_41188736369344
// MI455X (gfx1250) — matrix operand order repaired and checked by reading, not compiled, not run on hardware
//
#include <hip/hip_runtime.h>
#include <hip/hip_bf16.h>
#include <math.h>

typedef __bf16 v16bf __attribute__((ext_vector_type(16)));
typedef float  v8f   __attribute__((ext_vector_type(8)));

#define HIDDEN 2048
#define NH 32
#define NKV 4
#define HD 128
#define BB 2
#define SSEQ 2048
#define QKV_N ((NH + 2*NKV)*HD)   /* 5120 */
#define ROWS (BB*SSEQ)            /* 4096 */
#define LDA 40                    /* padded LDS row (bf16 elems) for 32-wide K tiles */

__device__ __forceinline__ v8f zero8() {
    v8f z;
#pragma unroll
    for (int i = 0; i < 8; ++i) z[i] = 0.0f;
    return z;
}

__device__ __forceinline__ v8f wmma_bf16(v16bf a, v16bf b, v8f c) {
    // D = A(16x32 bf16) * B(32x16 bf16) + C(16x16 f32)
    return __builtin_amdgcn_wmma_f32_16x16x32_bf16(false, a, false, b, (short)0, c, false, false);
}

// ---------------------------------------------------------------------------
// Kernel 1: qkv = hidden @ w_qkv^T, fused per-head RMSNorm + RoPE epilogue.
// Grid: (ROWS/128, QKV_N/128) = (32, 40). Block: 256 (8 waves, 4M x 2N).
// Each N tile of 128 == exactly one head.
// ---------------------------------------------------------------------------
__global__ __launch_bounds__(256) void qkv_rope_kernel(
    const float* __restrict__ hidden, const int* __restrict__ positions,
    const float* __restrict__ w_qkv, const float* __restrict__ q_norm_w,
    const float* __restrict__ k_norm_w,
    __bf16* __restrict__ q_bf, __bf16* __restrict__ k_bf, __bf16* __restrict__ v_bf)
{
    __shared__ __align__(16) char smem[65536];
    __bf16* abuf = (__bf16*)smem;       // [128][LDA]
    __bf16* bbuf = abuf + 128 * LDA;    // [128][LDA]
    float*  cbuf = (float*)smem;        // [128][128], reused after the k-loop

    const int tid   = threadIdx.x;
    const int lane  = tid & 31;
    const int wave  = tid >> 5;
    const int waveM = wave & 3;         // 0..3  -> 32 rows each
    const int waveN = wave >> 2;        // 0..1  -> 64 cols each
    const int tileM = blockIdx.x;
    const int tileN = blockIdx.y;
    const int lhalf = lane & 15;
    const bool hi   = (lane >= 16);

    v8f acc[2][4];
#pragma unroll
    for (int mt = 0; mt < 2; ++mt)
#pragma unroll
        for (int nt = 0; nt < 4; ++nt) acc[mt][nt] = zero8();

    const int srow = tid >> 1;          // 0..127
    const int scol = (tid & 1) * 16;    // 0 or 16

    const float* gArow = hidden + (size_t)(tileM * 128 + srow) * HIDDEN + scol;
    const float* gBrow = w_qkv  + (size_t)(tileN * 128 + srow) * HIDDEN + scol;

    for (int kk = 0; kk < HIDDEN; kk += 32) {
        // ---- stage A (activations) and B (weights) as bf16 into LDS ----
        {
            const float4* ga = (const float4*)(gArow + kk);
            const float4* gb = (const float4*)(gBrow + kk);
            __bf16* da = abuf + srow * LDA + scol;
            __bf16* db = bbuf + srow * LDA + scol;
#pragma unroll
            for (int v = 0; v < 4; ++v) {
                float4 fa = ga[v]; float4 fb = gb[v];
                da[v*4+0] = (__bf16)fa.x; da[v*4+1] = (__bf16)fa.y;
                da[v*4+2] = (__bf16)fa.z; da[v*4+3] = (__bf16)fa.w;
                db[v*4+0] = (__bf16)fb.x; db[v*4+1] = (__bf16)fb.y;
                db[v*4+2] = (__bf16)fb.z; db[v*4+3] = (__bf16)fb.w;
            }
            if (kk + 32 < HIDDEN) {             // prefetch next k tiles
                __builtin_prefetch(gArow + kk + 32, 0, 1);
                __builtin_prefetch(gBrow + kk + 32, 0, 1);
            }
        }
        __syncthreads();

        // ---- WMMA fragments from LDS ----
        v16bf afrag[2];
#pragma unroll
        for (int mt = 0; mt < 2; ++mt) {
            const __bf16* p = abuf + (waveM*32 + mt*16 + lhalf) * LDA + (hi ? 16 : 0);
            v16bf a;
#pragma unroll
            for (int i = 0; i < 8; ++i) { a[i] = p[i]; a[8+i] = p[8+i]; }
            afrag[mt] = a;
        }
#pragma unroll
        for (int nt = 0; nt < 4; ++nt) {
            const __bf16* p = bbuf + (waveN*64 + nt*16 + lhalf) * LDA + (hi ? 16 : 0);
            v16bf b;
#pragma unroll
            for (int i = 0; i < 16; ++i) b[i] = p[i];
#pragma unroll
            for (int mt = 0; mt < 2; ++mt)
                acc[mt][nt] = wmma_bf16(afrag[mt], b, acc[mt][nt]);
        }
        __syncthreads();
    }

    // ---- spill C tile to LDS for full-head-row epilogue ----
#pragma unroll
    for (int mt = 0; mt < 2; ++mt)
#pragma unroll
        for (int nt = 0; nt < 4; ++nt)
#pragma unroll
            for (int j = 0; j < 8; ++j) {
                int row = waveM*32 + mt*16 + (hi ? 8 : 0) + j;
                int col = waveN*64 + nt*16 + lhalf;
                cbuf[row * 128 + col] = acc[mt][nt][j];
            }
    __syncthreads();

    if (tid < 128) {
        const int row = tid;
        const size_t grow = (size_t)tileM * 128 + row;
        const int b = (int)(grow / SSEQ), s = (int)(grow % SSEQ);
        const int pos = positions[b * SSEQ + s];
        const float* c = cbuf + row * 128;

        if (tileN < NH + NKV) {
            const bool isQ = (tileN < NH);
            // RMS over the full 128-dim head (skewed reads to spread banks)
            float ssum = 0.0f;
            for (int i = 0; i < 128; ++i) { float x = c[(i + row) & 127]; ssum += x * x; }
            const float inv = rsqrtf(ssum * (1.0f / 128.0f) + 1e-6f);
            const float* nw = isQ ? q_norm_w : k_norm_w;
            __bf16* dst = isQ
                ? (q_bf + grow * (size_t)(NH * HD) + tileN * HD)
                : (k_bf + grow * (size_t)(NKV * HD) + (tileN - NH) * HD);
            // RoPE: ang = pos * theta^(-d/64), pair (d, d+64)
#pragma unroll 4
            for (int d = 0; d < 64; ++d) {
                float x1 = c[d]      * inv * nw[d];
                float x2 = c[d + 64] * inv * nw[d + 64];
                float ang = (float)pos * __expf(-0.21586735f * (float)d); // ln(1e6)/64
                float cs = __cosf(ang), sn = __sinf(ang);
                dst[d]      = (__bf16)(x1 * cs - x2 * sn);
                dst[d + 64] = (__bf16)(x2 * cs + x1 * sn);
            }
        } else {
            __bf16* dst = v_bf + grow * (size_t)(NKV * HD) + (tileN - NH - NKV) * HD;
#pragma unroll 8
            for (int d = 0; d < 128; ++d) dst[d] = (__bf16)c[d];
        }
    }
}

// ---------------------------------------------------------------------------
// Kernel 2: causal flash attention. Grid: (SSEQ/128, NH, BB). Block 256.
// Wave grid 8(M)x1(N): each wave owns 16 full score rows -> softmax stats
// reduce with shfl_xor within 16-lane halves (wave32).
// ---------------------------------------------------------------------------
#define KB 64
__global__ __launch_bounds__(256) void flash_attn_kernel(
    const __bf16* __restrict__ q_bf, const __bf16* __restrict__ k_bf,
    const __bf16* __restrict__ v_bf, __bf16* __restrict__ attn_bf)
{
    __shared__ __bf16 kbuf[64 * 136];   // K block  [k][d], padded
    __shared__ __bf16 vtbuf[128 * 72];  // V block transposed [d][k], padded
    __shared__ __bf16 pbuf[128 * 72];   // P (probs) [qrow][k], padded

    const int tid   = threadIdx.x;
    const int lane  = tid & 31;
    const int wave  = tid >> 5;
    const int lhalf = lane & 15;
    const bool hi   = (lane >= 16);
    const int qb = blockIdx.x, h = blockIdx.y, bz = blockIdx.z;
    const int kvh = h >> 3;             // rep = NH/NKV = 8

    // ---- Q fragments resident in registers (wave's 16 rows x 128 d) ----
    v16bf qfrag[4];
    {
        const size_t qrow = (size_t)bz * SSEQ + qb * 128 + wave * 16 + lhalf;
        const __bf16* qp = q_bf + qrow * (size_t)(NH * HD) + h * HD;
#pragma unroll
        for (int ks = 0; ks < 4; ++ks) {
            const __bf16* p = qp + ks * 32 + (hi ? 16 : 0);
            v16bf a;
#pragma unroll
            for (int i = 0; i < 8; ++i) { a[i] = p[i]; a[8+i] = p[8+i]; }
            qfrag[ks] = a;
        }
    }

    v8f oacc[8];
#pragma unroll
    for (int t = 0; t < 8; ++t) oacc[t] = zero8();
    float mrow[8], lrow[8];
#pragma unroll
    for (int j = 0; j < 8; ++j) { mrow[j] = -3.0e38f; lrow[j] = 0.0f; }

    const int nkb = (qb + 1) * 2;       // causal: k <= qb*128+127
    const float scale = 0.0883883476f;  // 1/sqrt(128)

    for (int kb = 0; kb < nkb; ++kb) {
        // ---- stage K (row-major) and V (transposed) for this 64-k block ----
        {
            const size_t base = ((size_t)bz * SSEQ + kb * KB) * (size_t)(NKV * HD) + kvh * HD;
            const __bf16* kg = k_bf + base;
            const __bf16* vg = v_bf + base;
#pragma unroll 4
            for (int i = 0; i < 32; ++i) {
                int idx = i * 256 + tid;        // 64*128 elements
                int r = idx >> 7, d = idx & 127;
                kbuf[r * 136 + d] = kg[(size_t)r * (NKV * HD) + d];
                vtbuf[d * 72 + r] = vg[(size_t)r * (NKV * HD) + d];
            }
        }
        __syncthreads();

        // ---- S = Q K^T (wave's 16 rows x 64 cols) ----
        v8f sacc[4];
#pragma unroll
        for (int nt = 0; nt < 4; ++nt) sacc[nt] = zero8();
#pragma unroll
        for (int ks = 0; ks < 4; ++ks)
#pragma unroll
            for (int nt = 0; nt < 4; ++nt) {
                const __bf16* p = kbuf + (nt*16 + lhalf) * 136 + ks*32 + (hi ? 16 : 0);
                v16bf b;
#pragma unroll
                for (int i = 0; i < 16; ++i) b[i] = p[i];
                sacc[nt] = wmma_bf16(qfrag[ks], b, sacc[nt]);
            }

        // ---- causal mask + online softmax (row-exclusive per wave) ----
        float pvals[4][8];
        const int qbase = qb * 128 + wave * 16 + (hi ? 8 : 0);
        const int kcol0 = kb * KB + lhalf;
#pragma unroll
        for (int j = 0; j < 8; ++j) {
            float mx = -3.0e38f;
#pragma unroll
            for (int nt = 0; nt < 4; ++nt) {
                float v = sacc[nt][j] * scale;
                if (kcol0 + nt * 16 > qbase + j) v = -3.0e38f;
                pvals[nt][j] = v;
                mx = fmaxf(mx, v);
            }
#pragma unroll
            for (int m = 8; m >= 1; m >>= 1) mx = fmaxf(mx, __shfl_xor(mx, m, 32));
            float mnew  = fmaxf(mrow[j], mx);
            float alpha = __expf(mrow[j] - mnew);
            float rsum  = 0.0f;
#pragma unroll
            for (int nt = 0; nt < 4; ++nt) {
                float p = __expf(pvals[nt][j] - mnew);
                pvals[nt][j] = p;
                rsum += p;
            }
#pragma unroll
            for (int m = 8; m >= 1; m >>= 1) rsum += __shfl_xor(rsum, m, 32);
            lrow[j] = lrow[j] * alpha + rsum;
            mrow[j] = mnew;
#pragma unroll
            for (int t = 0; t < 8; ++t) oacc[t][j] *= alpha;
        }

        // ---- P -> LDS (C-layout to A-layout conversion) ----
#pragma unroll
        for (int nt = 0; nt < 4; ++nt)
#pragma unroll
            for (int j = 0; j < 8; ++j)
                pbuf[(wave*16 + (hi ? 8 : 0) + j) * 72 + nt*16 + lhalf] = (__bf16)pvals[nt][j];
        __syncthreads();

        // ---- O += P V ----
#pragma unroll
        for (int ks = 0; ks < 2; ++ks) {
            const __bf16* pp = pbuf + (wave*16 + lhalf) * 72 + ks*32 + (hi ? 16 : 0);
            v16bf pa;
#pragma unroll
            for (int i = 0; i < 8; ++i) { pa[i] = pp[i]; pa[8+i] = pp[8+i]; }
#pragma unroll
            for (int dt = 0; dt < 8; ++dt) {
                const __bf16* vp = vtbuf + (dt*16 + lhalf) * 72 + ks*32 + (hi ? 16 : 0);
                v16bf b;
#pragma unroll
                for (int i = 0; i < 16; ++i) b[i] = vp[i];
                oacc[dt] = wmma_bf16(pa, b, oacc[dt]);
            }
        }
        __syncthreads();
    }

    // ---- normalize and store attn output (bf16) ----
    const size_t rowg = (size_t)bz * SSEQ + qb * 128 + wave * 16 + (hi ? 8 : 0);
#pragma unroll
    for (int dt = 0; dt < 8; ++dt)
#pragma unroll
        for (int j = 0; j < 8; ++j)
            attn_bf[(rowg + j) * (size_t)(NH * HD) + h * HD + dt*16 + lhalf] =
                (__bf16)(oacc[dt][j] / lrow[j]);
}

// ---------------------------------------------------------------------------
// Kernel 3: out = attn @ w_o^T. Grid: (ROWS/128, HIDDEN/128) = (32,16).
// ---------------------------------------------------------------------------
__global__ __launch_bounds__(256) void out_proj_kernel(
    const __bf16* __restrict__ attn_bf, const float* __restrict__ w_o,
    float* __restrict__ out)
{
    __shared__ __align__(16) __bf16 abuf[128 * LDA];
    __shared__ __align__(16) __bf16 bbuf[128 * LDA];

    const int tid   = threadIdx.x;
    const int lane  = tid & 31;
    const int wave  = tid >> 5;
    const int waveM = wave & 3;
    const int waveN = wave >> 2;
    const int tileM = blockIdx.x;
    const int tileN = blockIdx.y;
    const int lhalf = lane & 15;
    const bool hi   = (lane >= 16);

    v8f acc[2][4];
#pragma unroll
    for (int mt = 0; mt < 2; ++mt)
#pragma unroll
        for (int nt = 0; nt < 4; ++nt) acc[mt][nt] = zero8();

    const int srow = tid >> 1;
    const int scol = (tid & 1) * 16;
    const __bf16* gArow = attn_bf + (size_t)(tileM * 128 + srow) * (NH * HD) + scol;
    const float*  gBrow = w_o     + (size_t)(tileN * 128 + srow) * (NH * HD) + scol;

    for (int kk = 0; kk < NH * HD; kk += 32) {
        {   // A already bf16: raw 32-byte copy
            const uint4* ga = (const uint4*)(gArow + kk);
            uint4* da = (uint4*)(abuf + srow * LDA + scol);
            da[0] = ga[0]; da[1] = ga[1];
            // B: f32 -> bf16
            const float4* gb = (const float4*)(gBrow + kk);
            __bf16* db = bbuf + srow * LDA + scol;
#pragma unroll
            for (int v = 0; v < 4; ++v) {
                float4 fb = gb[v];
                db[v*4+0] = (__bf16)fb.x; db[v*4+1] = (__bf16)fb.y;
                db[v*4+2] = (__bf16)fb.z; db[v*4+3] = (__bf16)fb.w;
            }
            if (kk + 32 < NH * HD) {
                __builtin_prefetch(gArow + kk + 32, 0, 1);
                __builtin_prefetch(gBrow + kk + 32, 0, 1);
            }
        }
        __syncthreads();

        v16bf afrag[2];
#pragma unroll
        for (int mt = 0; mt < 2; ++mt) {
            const __bf16* p = abuf + (waveM*32 + mt*16 + lhalf) * LDA + (hi ? 16 : 0);
            v16bf a;
#pragma unroll
            for (int i = 0; i < 8; ++i) { a[i] = p[i]; a[8+i] = p[8+i]; }
            afrag[mt] = a;
        }
#pragma unroll
        for (int nt = 0; nt < 4; ++nt) {
            const __bf16* p = bbuf + (waveN*64 + nt*16 + lhalf) * LDA + (hi ? 16 : 0);
            v16bf b;
#pragma unroll
            for (int i = 0; i < 16; ++i) b[i] = p[i];
#pragma unroll
            for (int mt = 0; mt < 2; ++mt)
                acc[mt][nt] = wmma_bf16(afrag[mt], b, acc[mt][nt]);
        }
        __syncthreads();
    }

#pragma unroll
    for (int mt = 0; mt < 2; ++mt)
#pragma unroll
        for (int nt = 0; nt < 4; ++nt)
#pragma unroll
            for (int j = 0; j < 8; ++j) {
                size_t row = (size_t)tileM * 128 + waveM*32 + mt*16 + (hi ? 8 : 0) + j;
                int    col = tileN * 128 + waveN*64 + nt*16 + lhalf;
                out[row * HIDDEN + col] = acc[mt][nt][j];
            }
}

// ---------------------------------------------------------------------------
extern "C" void kernel_launch(void* const* d_in, const int* in_sizes, int n_in,
                              void* d_out, int out_size, void* d_ws, size_t ws_size,
                              hipStream_t stream)
{
    (void)in_sizes; (void)n_in; (void)out_size; (void)ws_size;

    const float* hidden    = (const float*)d_in[0];
    const int*   positions = (const int*)d_in[1];
    const float* w_qkv     = (const float*)d_in[2];
    const float* w_o       = (const float*)d_in[3];
    const float* q_norm_w  = (const float*)d_in[4];
    const float* k_norm_w  = (const float*)d_in[5];

    __bf16* q_bf    = (__bf16*)d_ws;                       // 4096*4096
    __bf16* k_bf    = q_bf + (size_t)ROWS * NH * HD;       // 4096*512
    __bf16* v_bf    = k_bf + (size_t)ROWS * NKV * HD;      // 4096*512
    __bf16* attn_bf = v_bf + (size_t)ROWS * NKV * HD;      // 4096*4096

    qkv_rope_kernel<<<dim3(ROWS / 128, QKV_N / 128), 256, 0, stream>>>(
        hidden, positions, w_qkv, q_norm_w, k_norm_w, q_bf, k_bf, v_bf);

    flash_attn_kernel<<<dim3(SSEQ / 128, NH, BB), 256, 0, stream>>>(
        q_bf, k_bf, v_bf, attn_bf);

    out_proj_kernel<<<dim3(ROWS / 128, HIDDEN / 128), 256, 0, stream>>>(
        attn_bf, w_o, (float*)d_out);
}
